// AdaptiveSoftmax_85942295593411
// MI455X (gfx1250) — matrix operand order repaired and checked by reading, not compiled, not run on hardware
//
#include <hip/hip_runtime.h>
#include <hip/hip_bf16.h>
#include <math.h>

typedef unsigned short ushort_t;
typedef __attribute__((ext_vector_type(16))) __bf16        v16bf;
typedef __attribute__((ext_vector_type(8)))  float         v8f;
typedef __attribute__((ext_vector_type(8)))  unsigned int  v8u;

#define NEG_INF (-__builtin_inff())

__device__ __forceinline__ ushort_t f2bf(float f) {
  unsigned int u = __float_as_uint(f);
  u += 0x7FFFu + ((u >> 16) & 1u);          // round-to-nearest-even
  return (ushort_t)(u >> 16);
}

// -------------------- conversion kernels --------------------
__global__ void k_f32_to_bf16(const float* __restrict__ in, ushort_t* __restrict__ out, long n) {
  long i = (long)blockIdx.x * blockDim.x + threadIdx.x;
  if (i < n) out[i] = f2bf(in[i]);
}

// in: fp32 [K x M] (row-major), out: bf16 [Mpad x K] (row-major transposed, zero-padded rows)
__global__ void k_transpose_bf16(const float* __restrict__ in, ushort_t* __restrict__ out,
                                 int K, int M, int Mpad) {
  long i = (long)blockIdx.x * blockDim.x + threadIdx.x;
  long total = (long)Mpad * K;
  if (i >= total) return;
  int m = (int)(i / K);
  int k = (int)(i - (long)m * K);
  float v = (m < M) ? in[(long)k * M + m] : 0.0f;
  out[i] = f2bf(v);
}

// -------------------- WMMA fragment loaders --------------------
// A 16x32 bf16, row-major source. Lane layout (ISA 7.12.2):
// lanes 0-15: row M=lane, regs 0-3 -> K=0..7, regs 4-7 -> K=16..23 (half 0)
// lanes 16-31: same rows, K shifted by +8.
__device__ __forceinline__ v16bf load_a_frag(const ushort_t* __restrict__ rowbase, int kb) {
  v8u u;
#pragma unroll
  for (int v = 0; v < 8; ++v)
    u[v] = *(const unsigned int*)(rowbase + kb + 2 * v + (v >= 4 ? 8 : 0));
  return __builtin_bit_cast(v16bf, u);
}

// B 32x16 bf16 fed from transposed weights Wt[M][K] (col n of W = row n of Wt).
// Lanes 0-15: col=lane, regs 0-3 -> K=0..7, regs 4-7 -> K=16..23; lanes 16-31: K shifted by +8.
__device__ __forceinline__ v16bf load_b_frag(const ushort_t* __restrict__ colbase, int kb) {
  v8u u;
#pragma unroll
  for (int v = 0; v < 8; ++v)
    u[v] = *(const unsigned int*)(colbase + kb + 2 * v + (v >= 4 ? 8 : 0));
  return __builtin_bit_cast(v16bf, u);
}

// -------------------- software-pipelined 16x64 GEMM core --------------------
// Accumulates a 16-token x 64-column tile over K (multiple of 32).
// Next k-step's A + 4 B fragments are issued BEFORE the current step's 4 WMMAs,
// so s_wait_loadcnt lands after a burst of matrix work instead of before each WMMA.
struct Acc4 { v8f c0, c1, c2, c3; };

__device__ __forceinline__ Acc4 gemm_16x64(const ushort_t* __restrict__ arow,
                                           const ushort_t* __restrict__ bcol,
                                           size_t ldw, int K) {
  Acc4 acc;
  acc.c0 = v8f{}; acc.c1 = v8f{}; acc.c2 = v8f{}; acc.c3 = v8f{};
  // prologue: first k-step fragments
  v16bf a  = load_a_frag(arow, 0);
  v16bf b0 = load_b_frag(bcol + 0 * 16 * ldw, 0);
  v16bf b1 = load_b_frag(bcol + 1 * 16 * ldw, 0);
  v16bf b2 = load_b_frag(bcol + 2 * 16 * ldw, 0);
  v16bf b3 = load_b_frag(bcol + 3 * 16 * ldw, 0);
  for (int kb = 32; kb < K; kb += 32) {
    // prefetch next k-step
    v16bf an  = load_a_frag(arow, kb);
    v16bf n0  = load_b_frag(bcol + 0 * 16 * ldw, kb);
    v16bf n1  = load_b_frag(bcol + 1 * 16 * ldw, kb);
    v16bf n2  = load_b_frag(bcol + 2 * 16 * ldw, kb);
    v16bf n3  = load_b_frag(bcol + 3 * 16 * ldw, kb);
    // compute current k-step
    acc.c0 = __builtin_amdgcn_wmma_f32_16x16x32_bf16(false, a, false, b0, (short)0, acc.c0, false, false);
    acc.c1 = __builtin_amdgcn_wmma_f32_16x16x32_bf16(false, a, false, b1, (short)0, acc.c1, false, false);
    acc.c2 = __builtin_amdgcn_wmma_f32_16x16x32_bf16(false, a, false, b2, (short)0, acc.c2, false, false);
    acc.c3 = __builtin_amdgcn_wmma_f32_16x16x32_bf16(false, a, false, b3, (short)0, acc.c3, false, false);
    a = an; b0 = n0; b1 = n1; b2 = n2; b3 = n3;
  }
  // epilogue: last k-step
  acc.c0 = __builtin_amdgcn_wmma_f32_16x16x32_bf16(false, a, false, b0, (short)0, acc.c0, false, false);
  acc.c1 = __builtin_amdgcn_wmma_f32_16x16x32_bf16(false, a, false, b1, (short)0, acc.c1, false, false);
  acc.c2 = __builtin_amdgcn_wmma_f32_16x16x32_bf16(false, a, false, b2, (short)0, acc.c2, false, false);
  acc.c3 = __builtin_amdgcn_wmma_f32_16x16x32_bf16(false, a, false, b3, (short)0, acc.c3, false, false);
  return acc;
}

// -------------------- GEMM + streaming CE partials --------------------
// One wave per block. Block computes a 16-token x 64-column tile of logits,
// emits per-row (max, sum exp(x-max)) partials and the label logit.
__global__ __launch_bounds__(32)
void k_gemm_ce(const ushort_t* __restrict__ A, int lda, int K,
               const ushort_t* __restrict__ Wt,
               const float* __restrict__ bias, int Mtot,
               const int* __restrict__ labels, int mode,
               float* __restrict__ pmax, float* __restrict__ psum,
               float* __restrict__ plab, int nchunks) {
  const int lane = threadIdx.x & 31;
  const int half = lane >> 4;
  const int r    = lane & 15;
  const int t0   = blockIdx.x * 16;
  const int chunk = blockIdx.y;
  const int colbase = chunk * 64;

  const ushort_t* arow = A + (size_t)(t0 + r) * lda + 8 * half;
  const ushort_t* bcol = Wt + (size_t)(colbase + r) * K + 8 * half;

  Acc4 acc = gemm_16x64(arow, bcol, (size_t)K, K);
  v8f c[4] = {acc.c0, acc.c1, acc.c2, acc.c3};

  // bias add + mask out padded columns
#pragma unroll
  for (int j = 0; j < 4; ++j) {
    int col = colbase + 16 * j + r;
    if (col < Mtot) {
      float bj = bias[col];
#pragma unroll
      for (int v = 0; v < 8; ++v) c[j][v] += bj;
    } else {
#pragma unroll
      for (int v = 0; v < 8; ++v) c[j][v] = NEG_INF;
    }
  }

  // per-row (max, sumexp) across the 64 columns + label logit capture
#pragma unroll
  for (int v = 0; v < 8; ++v) {
    int row = v + 8 * half;
    int token = t0 + row;
    float lm = fmaxf(fmaxf(c[0][v], c[1][v]), fmaxf(c[2][v], c[3][v]));
#pragma unroll
    for (int m = 8; m >= 1; m >>= 1) lm = fmaxf(lm, __shfl_xor(lm, m, 32));
    float ls = 0.0f;
#pragma unroll
    for (int j = 0; j < 4; ++j) ls += __expf(c[j][v] - lm);
#pragma unroll
    for (int m = 8; m >= 1; m >>= 1) ls += __shfl_xor(ls, m, 32);
    if (r == 0) {
      pmax[(size_t)token * nchunks + chunk] = lm;
      psum[(size_t)token * nchunks + chunk] = ls;
    }
    int l = labels[token];
    int ct;
    if (mode == 0) {
      ct = (l >= 28000) ? 16001 : ((l >= 16000) ? 16000 : l);
    } else if (mode == 1) {
      ct = l - 16000; if (ct < 0) ct = 0; if (ct > 11999) ct = 11999;
    } else {
      ct = l - 28000; if (ct < 0) ct = 0; if (ct > 7999) ct = 7999;
    }
#pragma unroll
    for (int j = 0; j < 4; ++j) {
      int col = colbase + 16 * j + r;
      if (col == ct) plab[token] = c[j][v];
    }
  }
}

// -------------------- GEMM -> bf16 activations (tail projections) --------------------
__global__ __launch_bounds__(32)
void k_gemm_proj(const ushort_t* __restrict__ A, int lda, int K,
                 const ushort_t* __restrict__ Wt,
                 const float* __restrict__ bias,
                 ushort_t* __restrict__ out, int ldo) {
  const int lane = threadIdx.x & 31;
  const int half = lane >> 4;
  const int r    = lane & 15;
  const int t0   = blockIdx.x * 16;
  const int colbase = blockIdx.y * 64;

  const ushort_t* arow = A + (size_t)(t0 + r) * lda + 8 * half;
  const ushort_t* bcol = Wt + (size_t)(colbase + r) * K + 8 * half;

  Acc4 acc = gemm_16x64(arow, bcol, (size_t)K, K);
  v8f c[4] = {acc.c0, acc.c1, acc.c2, acc.c3};

#pragma unroll
  for (int j = 0; j < 4; ++j) {
    int col = colbase + 16 * j + r;
    float bj = bias[col];
#pragma unroll
    for (int v = 0; v < 8; ++v) {
      int row = v + 8 * half;
      out[(size_t)(t0 + row) * ldo + col] = f2bf(c[j][v] + bj);
    }
  }
}

// -------------------- final reduce --------------------
__global__ void k_zero(float* out) { out[0] = 0.0f; }

__device__ __forceinline__ float lse_from_partials(const float* pm, const float* ps, int C) {
  float M = NEG_INF;
  for (int c = 0; c < C; ++c) M = fmaxf(M, pm[c]);
  float Z = 0.0f;
  for (int c = 0; c < C; ++c) Z += ps[c] * __expf(pm[c] - M);
  return M + __logf(Z);
}

__global__ void k_reduce_loss(const int* __restrict__ labels,
                              const float* __restrict__ hmax, const float* __restrict__ hsum,
                              const float* __restrict__ hlab, int HC,
                              const float* __restrict__ m0, const float* __restrict__ s0,
                              const float* __restrict__ lab0, int C0,
                              const float* __restrict__ m1, const float* __restrict__ s1,
                              const float* __restrict__ lab1, int C1,
                              float* __restrict__ out, int N) {
  int t = blockIdx.x * blockDim.x + threadIdx.x;
  if (t >= N) return;
  int l = labels[t];
  float loss = lse_from_partials(hmax + (size_t)t * HC, hsum + (size_t)t * HC, HC) - hlab[t];
  if (l >= 16000 && l < 28000)
    loss += lse_from_partials(m0 + (size_t)t * C0, s0 + (size_t)t * C0, C0) - lab0[t];
  else if (l >= 28000)
    loss += lse_from_partials(m1 + (size_t)t * C1, s1 + (size_t)t * C1, C1) - lab1[t];
  if (l == 0) loss = 0.0f;
  atomicAdd(out, loss * (1.0f / (float)N));
}

// -------------------- host launch --------------------
extern "C" void kernel_launch(void* const* d_in, const int* in_sizes, int n_in,
                              void* d_out, int out_size, void* d_ws, size_t ws_size,
                              hipStream_t stream) {
  (void)in_sizes; (void)n_in; (void)out_size; (void)ws_size;
  const float* inp    = (const float*)d_in[0];
  const int*   labels = (const int*)d_in[1];
  const float* head_W = (const float*)d_in[2];
  const float* head_b = (const float*)d_in[3];
  const float* t0_pW  = (const float*)d_in[4];
  const float* t0_pb  = (const float*)d_in[5];
  const float* t0_W   = (const float*)d_in[6];
  const float* t0_b   = (const float*)d_in[7];
  const float* t1_pW  = (const float*)d_in[8];
  const float* t1_pb  = (const float*)d_in[9];
  const float* t1_W   = (const float*)d_in[10];
  const float* t1_b   = (const float*)d_in[11];

  const int N = 4096, HID = 1024;
  const int HM = 16002, HMp = 16064, HC = HMp / 64;            // 251
  const int P0 = 256,  M0 = 12000, M0p = 12032, C0 = M0p / 64; // 188
  const int P1 = 64,   M1 = 8000,  M1p = 8000,  C1 = M1p / 64; // 125

  char* ws = (char*)d_ws;
  size_t off = 0;
  auto take = [&](size_t bytes) -> char* {
    char* p = ws + off;
    off = (off + bytes + 255) & ~(size_t)255;
    return p;
  };
  ushort_t* inp_bf = (ushort_t*)take((size_t)N * HID * 2);
  ushort_t* hWt    = (ushort_t*)take((size_t)HMp * HID * 2);
  ushort_t* p0Wt   = (ushort_t*)take((size_t)P0 * HID * 2);
  ushort_t* t0Wt   = (ushort_t*)take((size_t)M0p * P0 * 2);
  ushort_t* p1Wt   = (ushort_t*)take((size_t)P1 * HID * 2);
  ushort_t* t1Wt   = (ushort_t*)take((size_t)M1p * P1 * 2);
  ushort_t* proj0  = (ushort_t*)take((size_t)N * P0 * 2);
  ushort_t* proj1  = (ushort_t*)take((size_t)N * P1 * 2);
  float* hmax = (float*)take((size_t)N * HC * 4);
  float* hsum = (float*)take((size_t)N * HC * 4);
  float* hlab = (float*)take((size_t)N * 4);
  float* m0   = (float*)take((size_t)N * C0 * 4);
  float* s0   = (float*)take((size_t)N * C0 * 4);
  float* lab0 = (float*)take((size_t)N * 4);
  float* m1   = (float*)take((size_t)N * C1 * 4);
  float* s1   = (float*)take((size_t)N * C1 * 4);
  float* lab1 = (float*)take((size_t)N * 4);

  // 1) precision conversion + weight transposes (bf16, [M x K], zero padded)
  {
    long n = (long)N * HID;
    k_f32_to_bf16<<<dim3((unsigned)((n + 255) / 256)), dim3(256), 0, stream>>>(inp, inp_bf, n);
  }
  {
    long n = (long)HMp * HID;
    k_transpose_bf16<<<dim3((unsigned)((n + 255) / 256)), dim3(256), 0, stream>>>(head_W, hWt, HID, HM, HMp);
  }
  {
    long n = (long)P0 * HID;
    k_transpose_bf16<<<dim3((unsigned)((n + 255) / 256)), dim3(256), 0, stream>>>(t0_pW, p0Wt, HID, P0, P0);
  }
  {
    long n = (long)M0p * P0;
    k_transpose_bf16<<<dim3((unsigned)((n + 255) / 256)), dim3(256), 0, stream>>>(t0_W, t0Wt, P0, M0, M0p);
  }
  {
    long n = (long)P1 * HID;
    k_transpose_bf16<<<dim3((unsigned)((n + 255) / 256)), dim3(256), 0, stream>>>(t1_pW, p1Wt, HID, P1, P1);
  }
  {
    long n = (long)M1p * P1;
    k_transpose_bf16<<<dim3((unsigned)((n + 255) / 256)), dim3(256), 0, stream>>>(t1_W, t1Wt, P1, M1, M1p);
  }

  dim3 wave(32);
  // 2) head GEMM + CE partials
  k_gemm_ce<<<dim3(N / 16, HC), wave, 0, stream>>>(inp_bf, HID, HID, hWt, head_b, HM,
                                                   labels, 0, hmax, hsum, hlab, HC);
  // 3) tail 0: projection then tail GEMM + CE partials
  k_gemm_proj<<<dim3(N / 16, P0 / 64), wave, 0, stream>>>(inp_bf, HID, HID, p0Wt, t0_pb, proj0, P0);
  k_gemm_ce<<<dim3(N / 16, C0), wave, 0, stream>>>(proj0, P0, P0, t0Wt, t0_b, M0,
                                                   labels, 1, m0, s0, lab0, C0);
  // 4) tail 1
  k_gemm_proj<<<dim3(N / 16, P1 / 64), wave, 0, stream>>>(inp_bf, HID, HID, p1Wt, t1_pb, proj1, P1);
  k_gemm_ce<<<dim3(N / 16, C1), wave, 0, stream>>>(proj1, P1, P1, t1Wt, t1_b, M1,
                                                   labels, 2, m1, s1, lab1, C1);
  // 5) combine
  k_zero<<<dim3(1), dim3(1), 0, stream>>>((float*)d_out);
  k_reduce_loss<<<dim3((N + 255) / 256), dim3(256), 0, stream>>>(
      labels, hmax, hsum, hlab, HC, m0, s0, lab0, C0, m1, s1, lab1, C1, (float*)d_out, N);
}
